// DefinitionProbingLSTM_57011395887325
// MI455X (gfx1250) — matrix operand order repaired and checked by reading, not compiled, not run on hardware
//
#include <hip/hip_runtime.h>
#include <hip/hip_bf16.h>

// ---------------------------------------------------------------------------
// Types for CDNA5 WMMA (wave32): v_wmma_f32_16x16x32_bf16
// ---------------------------------------------------------------------------
typedef __attribute__((ext_vector_type(16))) __bf16         v16bf;
typedef __attribute__((ext_vector_type(8)))  float          v8f;
typedef __attribute__((ext_vector_type(8)))  unsigned short v8us;

#define DEV __device__ __forceinline__

DEV __bf16 f2bf(float f) {
  union { float f; unsigned u; } uf; uf.f = f;
  unsigned u = uf.u;
  unsigned r = (u + 0x7FFFu + ((u >> 16) & 1u)) >> 16;   // round-to-nearest-even
  union { unsigned short s; __bf16 b; } ub; ub.s = (unsigned short)r;
  return ub.b;
}

DEV v8f wmma_bf16(v16bf a, v16bf b, v8f c) {
  // (neg_a, A, neg_b, B, c_mod, C, reuse_a, reuse_b)
  return __builtin_amdgcn_wmma_f32_16x16x32_bf16(false, a, false, b, (short)0, c,
                                                 false, false);
}

// ---------------------------------------------------------------------------
// CDNA5 async global->LDS copy (ASYNCcnt path), per-lane B128 transfer.
// dst_lds: 32-bit LDS byte address; src: 64-bit global address.
// ---------------------------------------------------------------------------
DEV unsigned lds_off(const void* p) {
  // generic -> addrspace(3) cast strips the LDS aperture; AS3 pointers are
  // 32-bit LDS byte offsets on AMDGPU.
  return (unsigned)(size_t)(const __attribute__((address_space(3))) void*)p;
}

DEV void async_load_b128(unsigned dst_lds, const void* src) {
  asm volatile("global_load_async_to_lds_b128 %0, %1, off"
               :: "v"(dst_lds), "v"(src)
               : "memory");
}

DEV void wait_async0() {
  asm volatile("s_wait_asynccnt 0x0" ::: "memory");
}

// A fragment: 16x32 bf16 (MxK), row-major source, leading dim lda (elements).
// Lane l (g=l>>4): row M = m0 + (l&15); halves 0..7 hold K=k0+8g..+7,
// halves 8..15 hold K=k0+16+8g..+7  (per ISA 16-bit A layout).
DEV v16bf load_A(const __bf16* __restrict__ src, int lda, int m0, int k0, int lane) {
  int m = m0 + (lane & 15);
  int g = (lane >> 4) & 1;
  const __bf16* p = src + (size_t)m * lda + k0 + 8 * g;
  union { v16bf v; v8us h[2]; } u;
  u.h[0] = *reinterpret_cast<const v8us*>(p);        // 16B load
  u.h[1] = *reinterpret_cast<const v8us*>(p + 16);   // 16B load
  return u.v;
}

// B fragment: 32x16 bf16 (KxN) taken as W^T, where W is row-major [N, K].
// Lane l: column N = n0 + (l&15); halves 0..7 hold K = k0 + 8*(l>>4) + j,
// halves 8..15 hold K = k0 + 16 + 8*(l>>4) + (j-8).
DEV v16bf load_B(const __bf16* __restrict__ W, int ldw, int n0, int k0, int lane) {
  int n = n0 + (lane & 15);
  int g = (lane >> 4) & 1;
  const __bf16* p = W + (size_t)n * ldw + k0 + 8 * g;
  union { v16bf v; v8us h[2]; } u;
  u.h[0] = *reinterpret_cast<const v8us*>(p);
  u.h[1] = *reinterpret_cast<const v8us*>(p + 16);
  return u.v;
}

// ---------------------------------------------------------------------------
// f32 -> bf16 conversion
// ---------------------------------------------------------------------------
__global__ void cvt_f32_bf16(const float* __restrict__ src, __bf16* __restrict__ dst,
                             long n) {
  long i = (long)blockIdx.x * blockDim.x + threadIdx.x;
  if (i < n) dst[i] = f2bf(src[i]);
}

// ---------------------------------------------------------------------------
// Embedding gather: x_bf[t][b][e] = bf16(emb[input_ids[b][t]][e]),  t in 0..62
// ---------------------------------------------------------------------------
__global__ void gather_embed(const int* __restrict__ ids,     // (64,64)
                             const float* __restrict__ emb,   // (32000,512)
                             __bf16* __restrict__ x_bf) {     // (63,64,512)
  int row = blockIdx.x;            // t*64 + b
  int t = row >> 6, b = row & 63;
  int id = ids[b * 64 + t];
  const float* src = emb + (size_t)id * 512;
  __bf16* dst = x_bf + (size_t)row * 512;
  for (int e = threadIdx.x; e < 512; e += blockDim.x) dst[e] = f2bf(src[e]);
}

// ---------------------------------------------------------------------------
// Generic D = A @ W^T GEMM, wave tile 16(M) x 32(N). Used for encoder proj.
// ---------------------------------------------------------------------------
__global__ void gemm_bf16_AWt(const __bf16* __restrict__ A, int lda,
                              const __bf16* __restrict__ W, int ldw,
                              int Ntiles, int K,
                              float* __restrict__ Df32, int ldd) {
  int lane = threadIdx.x & 31;
  int gw = (blockIdx.x * blockDim.x + threadIdx.x) >> 5;
  int mt = gw / Ntiles, nt = gw % Ntiles;
  int m0 = mt * 16, n0 = nt * 32;
  v8f acc0 = {}, acc1 = {};
  for (int k = 0; k < K; k += 32) {
    v16bf a = load_A(A, lda, m0, k, lane);
    acc0 = wmma_bf16(a, load_B(W, ldw, n0, k, lane), acc0);
    acc1 = wmma_bf16(a, load_B(W, ldw, n0 + 16, k, lane), acc1);
  }
  int n = lane & 15;
  int mrow = m0 + 8 * ((lane >> 4) & 1);   // C/D layout: M = r + 8*(lane>=16)
#pragma unroll
  for (int r = 0; r < 8; ++r) {
    Df32[(size_t)(mrow + r) * ldd + n0 + n]      = acc0[r];
    Df32[(size_t)(mrow + r) * ldd + n0 + 16 + n] = acc1[r];
  }
}

// ---------------------------------------------------------------------------
// Replicate encoder state: carry0 = (h0, h0, h0, h0)
// ---------------------------------------------------------------------------
__global__ void init_state(const float* __restrict__ h_init, float* __restrict__ c0,
                           float* __restrict__ c1, __bf16* __restrict__ h0_bf,
                           __bf16* __restrict__ h_all0) {
  int i = blockIdx.x * blockDim.x + threadIdx.x;   // 0..65535
  float v = h_init[i];
  c0[i] = v; c1[i] = v;
  __bf16 bv = f2bf(v);
  h0_bf[i] = bv; h_all0[i] = bv;
}

// ---------------------------------------------------------------------------
// LSTM gate GEMM: gates(64x4096) = X(64xKx) @ Wih^T + Hp(64xKh) @ Whh^T
// Latency-bound recurrent step: direct-from-L2, wave tile 16x32, 512 waves.
// ---------------------------------------------------------------------------
__global__ void lstm_gates(const __bf16* __restrict__ X, int Kx,
                           const __bf16* __restrict__ Hp, int Kh,
                           const __bf16* __restrict__ Wih,
                           const __bf16* __restrict__ Whh,
                           float* __restrict__ gates) {
  int lane = threadIdx.x & 31;
  int gw = (blockIdx.x * blockDim.x + threadIdx.x) >> 5;
  int mt = gw >> 7;          // 4096/32 = 128 N tiles
  int nt = gw & 127;
  int m0 = mt * 16, n0 = nt * 32;
  v8f acc0 = {}, acc1 = {};
  for (int k = 0; k < Kx; k += 32) {
    v16bf a = load_A(X, Kx, m0, k, lane);
    acc0 = wmma_bf16(a, load_B(Wih, Kx, n0, k, lane), acc0);
    acc1 = wmma_bf16(a, load_B(Wih, Kx, n0 + 16, k, lane), acc1);
  }
  for (int k = 0; k < Kh; k += 32) {
    v16bf a = load_A(Hp, Kh, m0, k, lane);
    acc0 = wmma_bf16(a, load_B(Whh, Kh, n0, k, lane), acc0);
    acc1 = wmma_bf16(a, load_B(Whh, Kh, n0 + 16, k, lane), acc1);
  }
  int n = lane & 15;
  int mrow = m0 + 8 * ((lane >> 4) & 1);
#pragma unroll
  for (int r = 0; r < 8; ++r) {
    gates[(size_t)(mrow + r) * 4096 + n0 + n]      = acc0[r];
    gates[(size_t)(mrow + r) * 4096 + n0 + 16 + n] = acc1[r];
  }
}

// ---------------------------------------------------------------------------
// LSTM pointwise: c_new = sig(f)*c + sig(i)*tanh(g); h = sig(o)*tanh(c_new)
// ---------------------------------------------------------------------------
__global__ void lstm_pointwise(const float* __restrict__ gates,
                               const float* __restrict__ b_ih,
                               const float* __restrict__ b_hh,
                               float* __restrict__ c,
                               __bf16* __restrict__ h_bf) {
  int idx = blockIdx.x * blockDim.x + threadIdx.x;  // 0..65535
  int b = idx >> 10, j = idx & 1023;
  const float* g = gates + (size_t)b * 4096;
  float gi = g[j]        + b_ih[j]        + b_hh[j];
  float gf = g[1024 + j] + b_ih[1024 + j] + b_hh[1024 + j];
  float gg = g[2048 + j] + b_ih[2048 + j] + b_hh[2048 + j];
  float go = g[3072 + j] + b_ih[3072 + j] + b_hh[3072 + j];
  float si = 1.f / (1.f + __expf(-gi));
  float sf = 1.f / (1.f + __expf(-gf));
  float so = 1.f / (1.f + __expf(-go));
  float cn = sf * c[idx] + si * tanhf(gg);
  float hn = so * tanhf(cn);
  c[idx] = cn;
  h_bf[idx] = f2bf(hn);
}

// ---------------------------------------------------------------------------
// Batched output projection over all 63 timesteps:
// logits(4032 x 32000) = Hseq(4032 x 1024) @ Wp^T + bias, remapped to [b][t][V].
//
// L2-bandwidth-bound kernel -> LDS-staged, double-buffered pipeline using
// CDNA5 async global->LDS copies (ASYNCcnt). Per K-step a block stages
// A: 64x32 bf16 (4 KB) and B: 256x32 bf16 (16 KB) exactly once, removing the
// 4x (A) / 2x (B) intra-block L2 redundancy of the direct version.
// Issue async for stage s+1 -> 8 WMMAs/wave from stage s (ds_load_b128
// fragment reads) -> s_wait_asynccnt 0 -> barrier. 40 KB LDS / 320 KB WGP.
// ---------------------------------------------------------------------------
__global__ void proj_gemm(const __bf16* __restrict__ Hseq,  // rows = t*64+b
                          const __bf16* __restrict__ Wp,    // (32000,1024)
                          const float*  __restrict__ bias,  // (32000)
                          float* __restrict__ logits) {     // [b][t][V]
  __shared__ __align__(16) __bf16 sA[2][64 * 32];
  __shared__ __align__(16) __bf16 sB[2][256 * 32];

  int tid = threadIdx.x;
  int lane = tid & 31;
  int w = tid >> 5;              // 0..7
  int wm = w >> 2, wn = w & 3;   // 2 x 4 wave grid
  int m_blk = blockIdx.y * 64;
  int n_blk = blockIdx.x * 256;

  // Stage one 32-wide K block into LDS buffer s (all 256 threads cooperate).
  auto stage = [&](int k, int s) {
    {
      int r = tid >> 2, c = (tid & 3) * 8;               // 256 x 16B chunks
      async_load_b128(lds_off(&sA[s][r * 32 + c]),
                      Hseq + (size_t)(m_blk + r) * 1024 + k + c);
    }
#pragma unroll
    for (int j = 0; j < 4; ++j) {                        // 1024 x 16B chunks
      int idx = tid + 256 * j;
      int r = idx >> 2, c = (idx & 3) * 8;
      async_load_b128(lds_off(&sB[s][r * 32 + c]),
                      Wp + (size_t)(n_blk + r) * 1024 + k + c);
    }
  };

  v8f acc[2][4];
#pragma unroll
  for (int i = 0; i < 2; ++i)
#pragma unroll
    for (int j = 0; j < 4; ++j) acc[i][j] = (v8f){};

  stage(0, 0);
  wait_async0();
  __syncthreads();

  int s = 0;
  for (int kt = 0; kt < 32; ++kt) {
    if (kt + 1 < 32) stage((kt + 1) * 32, s ^ 1);        // prefetch next stage

    // Compute out of LDS stage s (fragment reads are 16B LDS loads).
    v16bf a0 = load_A(&sA[s][0], 32, wm * 32,      0, lane);
    v16bf a1 = load_A(&sA[s][0], 32, wm * 32 + 16, 0, lane);
#pragma unroll
    for (int j = 0; j < 4; ++j) {
      v16bf b = load_B(&sB[s][0], 32, wn * 64 + 16 * j, 0, lane);
      acc[0][j] = wmma_bf16(a0, b, acc[0][j]);
      acc[1][j] = wmma_bf16(a1, b, acc[1][j]);
    }

    wait_async0();       // next stage's async copies (issued above) complete
    __syncthreads();     // all waves done writing s^1 and done reading s
    s ^= 1;
  }

  int n = lane & 15, g = (lane >> 4) & 1;
#pragma unroll
  for (int i = 0; i < 2; ++i) {
#pragma unroll
    for (int j = 0; j < 4; ++j) {
#pragma unroll
      for (int r = 0; r < 8; ++r) {
        int m = m_blk + wm * 32 + 16 * i + 8 * g + r;   // sequence row = t*64+b
        int t = m >> 6, bb = m & 63;
        int col = n_blk + wn * 64 + 16 * j + n;
        logits[(size_t)(bb * 63 + t) * 32000 + col] = acc[i][j][r] + bias[col];
      }
    }
  }
}

// ---------------------------------------------------------------------------
// Row-wise argmax over V=32000, tie-break to lowest index (matches jnp.argmax)
// ---------------------------------------------------------------------------
__global__ void argmax_rows(const float* __restrict__ logits, int* __restrict__ preds) {
  int m = blockIdx.x;                            // flat [b][t] row, matches preds
  const float* row = logits + (size_t)m * 32000;
  float best = -3.402823466e38f; int bi = 0;
  for (int v = threadIdx.x; v < 32000; v += blockDim.x) {
    float x = row[v];
    if (x > best) { best = x; bi = v; }
  }
  __shared__ float smax[256];
  __shared__ int   sidx[256];
  smax[threadIdx.x] = best; sidx[threadIdx.x] = bi;
  __syncthreads();
  for (int s = 128; s > 0; s >>= 1) {
    if (threadIdx.x < s) {
      float om = smax[threadIdx.x + s]; int oi = sidx[threadIdx.x + s];
      if (om > smax[threadIdx.x] || (om == smax[threadIdx.x] && oi < sidx[threadIdx.x])) {
        smax[threadIdx.x] = om; sidx[threadIdx.x] = oi;
      }
    }
    __syncthreads();
  }
  if (threadIdx.x == 0) preds[m] = sidx[0];
}

// ---------------------------------------------------------------------------
// Host launcher
// ---------------------------------------------------------------------------
extern "C" void kernel_launch(void* const* d_in, const int* in_sizes, int n_in,
                              void* d_out, int out_size, void* d_ws, size_t ws_size,
                              hipStream_t stream) {
  (void)in_sizes; (void)n_in; (void)out_size; (void)ws_size;

  const int*   ids          = (const int*)  d_in[0];   // (64,64)
  const float* init_state_f = (const float*)d_in[1];   // (64,2048)
  const float* emb          = (const float*)d_in[2];   // (32000,512)
  const float* enc_proj_W   = (const float*)d_in[3];   // (1024,2048)
  const float* W_ih0        = (const float*)d_in[4];   // (4096,512)
  const float* W_hh0        = (const float*)d_in[5];   // (4096,1024)
  const float* b_ih0        = (const float*)d_in[6];
  const float* b_hh0        = (const float*)d_in[7];
  const float* W_ih1        = (const float*)d_in[8];   // (4096,1024)
  const float* W_hh1        = (const float*)d_in[9];   // (4096,1024)
  const float* b_ih1        = (const float*)d_in[10];
  const float* b_hh1        = (const float*)d_in[11];
  const float* proj_W       = (const float*)d_in[12];  // (32000,1024)
  const float* proj_b       = (const float*)d_in[13];  // (32000)

  // Workspace carve-up (256B aligned)
  char* wsp = (char*)d_ws;
  size_t off = 0;
  auto ALLOC = [&](size_t bytes) -> void* {
    void* p = wsp + off;
    off += (bytes + 255) & ~(size_t)255;
    return p;
  };
  __bf16* is_bf   = (__bf16*)ALLOC((size_t)64 * 2048 * 2);
  __bf16* enc_bf  = (__bf16*)ALLOC((size_t)1024 * 2048 * 2);
  __bf16* wih0_bf = (__bf16*)ALLOC((size_t)4096 * 512 * 2);
  __bf16* whh0_bf = (__bf16*)ALLOC((size_t)4096 * 1024 * 2);
  __bf16* wih1_bf = (__bf16*)ALLOC((size_t)4096 * 1024 * 2);
  __bf16* whh1_bf = (__bf16*)ALLOC((size_t)4096 * 1024 * 2);
  __bf16* wp_bf   = (__bf16*)ALLOC((size_t)32000 * 1024 * 2);
  __bf16* x_bf    = (__bf16*)ALLOC((size_t)63 * 64 * 512 * 2);
  __bf16* h_all   = (__bf16*)ALLOC((size_t)64 * 64 * 1024 * 2); // slot 0 = init
  __bf16* h0_bf   = (__bf16*)ALLOC((size_t)64 * 1024 * 2);
  float*  h_init  = (float*) ALLOC((size_t)64 * 1024 * 4);
  float*  c0      = (float*) ALLOC((size_t)64 * 1024 * 4);
  float*  c1      = (float*) ALLOC((size_t)64 * 1024 * 4);
  float*  gates   = (float*) ALLOC((size_t)64 * 4096 * 4);

  auto cvt = [&](const float* s, __bf16* d, long n) {
    cvt_f32_bf16<<<(unsigned)((n + 255) / 256), 256, 0, stream>>>(s, d, n);
  };
  cvt(init_state_f, is_bf,   (long)64 * 2048);
  cvt(enc_proj_W,   enc_bf,  (long)1024 * 2048);
  cvt(W_ih0,        wih0_bf, (long)4096 * 512);
  cvt(W_hh0,        whh0_bf, (long)4096 * 1024);
  cvt(W_ih1,        wih1_bf, (long)4096 * 1024);
  cvt(W_hh1,        whh1_bf, (long)4096 * 1024);
  cvt(proj_W,       wp_bf,   (long)32000 * 1024);

  gather_embed<<<63 * 64, 256, 0, stream>>>(ids, emb, x_bf);

  // h0 = initial_state @ enc_proj_W^T  (M=64, N=1024, K=2048): 128 wave tiles
  gemm_bf16_AWt<<<16, 256, 0, stream>>>(is_bf, 2048, enc_bf, 2048,
                                        /*Ntiles=*/32, /*K=*/2048, h_init, 1024);
  init_state<<<256, 256, 0, stream>>>(h_init, c0, c1, h0_bf, h_all);

  // Sequential recurrence: 63 steps x 2 layers
  for (int t = 0; t < 63; ++t) {
    lstm_gates<<<64, 256, 0, stream>>>(x_bf + (size_t)t * 64 * 512, 512,
                                       h0_bf, 1024, wih0_bf, whh0_bf, gates);
    lstm_pointwise<<<256, 256, 0, stream>>>(gates, b_ih0, b_hh0, c0, h0_bf);

    lstm_gates<<<64, 256, 0, stream>>>(h0_bf, 1024,
                                       h_all + (size_t)t * 64 * 1024, 1024,
                                       wih1_bf, whh1_bf, gates);
    lstm_pointwise<<<256, 256, 0, stream>>>(gates, b_ih1, b_hh1, c1,
                                            h_all + (size_t)(t + 1) * 64 * 1024);
  }

  // Batched projection over all timesteps + bias, then argmax
  float* logits = (float*)d_out + 4032;                 // after preds region
  proj_gemm<<<dim3(125, 63), 256, 0, stream>>>(h_all + (size_t)64 * 1024,
                                               wp_bf, proj_b, logits);
  argmax_rows<<<4032, 256, 0, stream>>>(logits, (int*)d_out);
}
